// MyBert_58188216926298
// MI455X (gfx1250) — matrix operand order repaired and checked by reading, not compiled, not run on hardware
//
#include <hip/hip_runtime.h>
#include <hip/hip_bf16.h>

// ---------------------------------------------------------------------------
// BERT-base (L=12,H=768,NH=12,HD=64,FF=3072) B=16,S=512 on gfx1250 (MI455X).
// All GEMMs + attention einsums use v_wmma_f32_16x16x32_bf16 (wave32 WMMA).
// GEMM tiles staged with global_load_async_to_lds_b128 (ASYNCcnt), double
// buffered so DMA overlaps WMMA compute.
// ---------------------------------------------------------------------------

#define LAYERS  12
#define H       768
#define NH      12
#define HD      64
#define FF      3072
#define SEQ     512
#define BATCH   16
#define MROWS   (BATCH*SEQ)   // 8192
#define MMAP    256
#define NINTENT 32
#define NSLOT   128

typedef __attribute__((ext_vector_type(16))) __bf16 v16bf;
typedef __attribute__((ext_vector_type(8)))  float  v8f;

union ABfrag { v16bf v; unsigned int u[8]; uint4 q[2]; };

static __device__ __forceinline__ unsigned short f32_to_bf16(float f){
  union { float f; unsigned int u; } x; x.f = f;
  unsigned int r = x.u + 0x7FFFu + ((x.u >> 16) & 1u);   // round-to-nearest-even
  return (unsigned short)(r >> 16);
}

static __device__ __forceinline__ v8f wmma_bf16(const ABfrag& a, const ABfrag& b, v8f c){
  // D = A(16x32 bf16) x B(32x16 bf16) + C(16x16 f32)
  return __builtin_amdgcn_wmma_f32_16x16x32_bf16(false, a.v, false, b.v, (short)0, c, false, false);
}

// CDNA5 async DMA: global -> LDS, 16B per lane, tracked by ASYNCcnt.
static __device__ __forceinline__ void async_b128(const unsigned short* g, unsigned short* l){
  const unsigned       ldsa = (unsigned)(unsigned long long)l;  // LDS addr = low 32 bits
  const unsigned long long ga = (unsigned long long)g;
  asm volatile("global_load_async_to_lds_b128 %0, %1, off"
               :: "v"(ldsa), "v"(ga) : "memory");
}
static __device__ __forceinline__ void wait_async0(){
  asm volatile("s_wait_asynccnt 0x0" ::: "memory");
}

// block-wide sum of two values (256 threads = 8 wave32)
static __device__ __forceinline__ void block_reduce_2(float& s, float& q){
  #pragma unroll
  for (int off = 16; off > 0; off >>= 1){
    s += __shfl_xor(s, off, 32);
    q += __shfl_xor(q, off, 32);
  }
  __shared__ float shs[8], shq[8];
  const int w = threadIdx.x >> 5, l = threadIdx.x & 31;
  if (l == 0){ shs[w] = s; shq[w] = q; }
  __syncthreads();
  if (threadIdx.x == 0){
    float ts = 0.f, tq = 0.f;
    #pragma unroll
    for (int i = 0; i < 8; ++i){ ts += shs[i]; tq += shq[i]; }
    shs[0] = ts; shq[0] = tq;
  }
  __syncthreads();
  s = shs[0]; q = shq[0];
}

// ---------------------------------------------------------------------------
// Embeddings + LayerNorm -> x (fp32) and x (bf16)
// ---------------------------------------------------------------------------
__global__ __launch_bounds__(256) void k_embed_ln(
    const int* __restrict__ ids, const float* __restrict__ wemb,
    const float* __restrict__ pemb, const float* __restrict__ temb,
    const float* __restrict__ g, const float* __restrict__ bt,
    float* __restrict__ xf, unsigned short* __restrict__ xb)
{
  const int row = blockIdx.x;
  const int s   = row & (SEQ - 1);
  const int tid = threadIdx.x;
  const float* wr = wemb + (size_t)ids[row] * H;
  const float* pr = pemb + (size_t)s * H;
  float v[3]; float sum = 0.f, sq = 0.f;
  #pragma unroll
  for (int j = 0; j < 3; ++j){
    const int i = tid + j*256;
    const float t = wr[i] + pr[i] + temb[i];
    v[j] = t; sum += t; sq += t*t;
  }
  block_reduce_2(sum, sq);
  const float mean = sum * (1.f/H);
  const float var  = sq * (1.f/H) - mean*mean;
  const float inv  = rsqrtf(var + 1e-12f);
  #pragma unroll
  for (int j = 0; j < 3; ++j){
    const int i = tid + j*256;
    const float o = (v[j]-mean)*inv*g[i] + bt[i];
    xf[(size_t)row*H + i] = o;
    xb[(size_t)row*H + i] = f32_to_bf16(o);
  }
}

// x = LN(x + y); writes fp32 + bf16 (in-place on x is safe: row-local)
__global__ __launch_bounds__(256) void k_resid_ln(
    const float* __restrict__ xin, const float* __restrict__ y,
    const float* __restrict__ g, const float* __restrict__ bt,
    float* __restrict__ xf, unsigned short* __restrict__ xb)
{
  const int row = blockIdx.x;
  const int tid = threadIdx.x;
  float v[3]; float sum = 0.f, sq = 0.f;
  #pragma unroll
  for (int j = 0; j < 3; ++j){
    const int i = tid + j*256;
    const float t = xin[(size_t)row*H + i] + y[(size_t)row*H + i];
    v[j] = t; sum += t; sq += t*t;
  }
  block_reduce_2(sum, sq);
  const float mean = sum * (1.f/H);
  const float var  = sq * (1.f/H) - mean*mean;
  const float inv  = rsqrtf(var + 1e-12f);
  #pragma unroll
  for (int j = 0; j < 3; ++j){
    const int i = tid + j*256;
    const float o = (v[j]-mean)*inv*g[i] + bt[i];
    xf[(size_t)row*H + i] = o;
    xb[(size_t)row*H + i] = f32_to_bf16(o);
  }
}

// W[K][N] fp32 -> Wt[N][K] bf16
__global__ __launch_bounds__(256) void k_transpose_bf16(
    const float* __restrict__ w, unsigned short* __restrict__ out, int K, int N)
{
  const int idx = blockIdx.x*256 + threadIdx.x;
  if (idx >= K*N) return;
  const int k = idx / N, n = idx - k*N;
  out[(size_t)n*K + k] = f32_to_bf16(w[idx]);
}

// ---------------------------------------------------------------------------
// bf16 WMMA GEMM: C[M,N] = A[M,K] x Wt[N,K]^T + bias, fused epilogue.
// block = 256 thr (8 waves), tile 128x128, wave = 32x64 (2x4 WMMA tiles).
// LDS double-buffered via global_load_async_to_lds_b128.
// ---------------------------------------------------------------------------
enum { EPI_BF16 = 0, EPI_GELU = 1, EPI_F32 = 2 };

template <int EPI>
__global__ __launch_bounds__(256) void k_gemm(
    const unsigned short* __restrict__ A,
    const unsigned short* __restrict__ Wt,
    const float* __restrict__ bias,
    unsigned short* __restrict__ obf,
    float* __restrict__ of32,
    int N, int K)
{
  __shared__ __align__(16) unsigned short As[2][128*32];
  __shared__ __align__(16) unsigned short Bs[2][128*32];
  const int tid  = threadIdx.x;
  const int wave = tid >> 5, lane = tid & 31;
  const int wm   = wave >> 1, wn = wave & 1;          // 4 (M) x 2 (N) waves
  const int m0   = blockIdx.y * 128, n0 = blockIdx.x * 128;

  // stage one 128x32 A tile + one 128x32 B tile into LDS buffer `buf`
  auto stage = [&](int buf, int k0){
    #pragma unroll
    for (int it = 0; it < 2; ++it){
      const int idx = tid + it*256;
      const int row = idx >> 2, off = (idx & 3) * 8;  // 8 bf16 = one b128
      async_b128(&A [(size_t)(m0+row)*K + k0 + off], &As[buf][row*32 + off]);
      async_b128(&Wt[(size_t)(n0+row)*K + k0 + off], &Bs[buf][row*32 + off]);
    }
  };

  v8f acc[2][4] = {};
  stage(0, 0);
  wait_async0();
  __syncthreads();

  for (int k0 = 0; k0 < K; k0 += 32){
    const int cur = (k0 >> 5) & 1;
    if (k0 + 32 < K) stage(cur ^ 1, k0 + 32);         // DMA next tile during compute

    ABfrag af[2];
    #pragma unroll
    for (int mi = 0; mi < 2; ++mi){                   // A frag 16x32 (contiguous k = 16h + s)
      const int m = wm*32 + mi*16 + (lane & 15);
      #pragma unroll
      for (int j = 0; j < 8; ++j){
        const int k = ((lane >> 4) << 4) + (j << 1);
        af[mi].u[j] = *(const unsigned int*)&As[cur][m*32 + k];
      }
    }
    #pragma unroll
    for (int ni = 0; ni < 4; ++ni){                   // B frag 32x16 (contiguous k = 16h + s)
      ABfrag bfr;
      const int n  = wn*64 + ni*16 + (lane & 15);
      const int ks = (lane >> 4) << 4;
      const uint4* p = (const uint4*)&Bs[cur][n*32 + ks];
      bfr.q[0] = p[0]; bfr.q[1] = p[1];
      #pragma unroll
      for (int mi = 0; mi < 2; ++mi)
        acc[mi][ni] = wmma_bf16(af[mi], bfr, acc[mi][ni]);
    }
    wait_async0();
    __syncthreads();
  }

  #pragma unroll
  for (int mi = 0; mi < 2; ++mi){
    #pragma unroll
    for (int ni = 0; ni < 4; ++ni){
      const int n = n0 + wn*64 + ni*16 + (lane & 15);
      const float bv = bias ? bias[n] : 0.f;
      #pragma unroll
      for (int r = 0; r < 8; ++r){
        const int m = m0 + wm*32 + mi*16 + ((lane >> 4) << 3) + r;
        float vv = acc[mi][ni][r] + bv;
        if (EPI == EPI_GELU) vv = 0.5f * vv * (1.f + erff(vv * 0.70710678118f));
        if (EPI == EPI_F32) of32[(size_t)m*N + n] = vv;
        else                obf[(size_t)m*N + n] = f32_to_bf16(vv);
      }
    }
  }
}

// ---------------------------------------------------------------------------
// Flash attention: block = (b, h, 128 q-rows); wave = 16 q-rows.
// scores = Q K^T * scale + maskbias; online softmax; O = P V.
// ---------------------------------------------------------------------------
__global__ __launch_bounds__(256) void k_attention(
    const unsigned short* __restrict__ qkv,  // [MROWS, 3H] bf16 (q|k|v)
    const int* __restrict__ amask,           // [BATCH, SEQ]
    unsigned short* __restrict__ ctx)        // [MROWS, H] bf16
{
  __shared__ __align__(16) unsigned short Qs[128*64];
  __shared__ __align__(16) unsigned short Ks[64*64];   // [key][d]
  __shared__ __align__(16) unsigned short Vs[64*64];   // [d][key] (transposed)
  __shared__ __align__(16) unsigned short Ps[8][16*64];

  const int qblk = blockIdx.x;
  const int h    = blockIdx.y;
  const int b    = blockIdx.z;
  const int tid  = threadIdx.x;
  const int wave = tid >> 5, lane = tid & 31;
  const size_t RS = 3*H;

  { // stage Q tile 128x64
    const int q = tid >> 1;
    const int doff = (tid & 1) * 32;
    const unsigned short* src = qkv + (size_t)(b*SEQ + qblk*128 + q)*RS + h*HD + doff;
    uint4* dst = (uint4*)&Qs[q*64 + doff];
    const uint4* s4 = (const uint4*)src;
    dst[0]=s4[0]; dst[1]=s4[1]; dst[2]=s4[2]; dst[3]=s4[3];
  }
  __syncthreads();

  ABfrag aq[2];                         // Q A-frags, invariant over key blocks
  #pragma unroll
  for (int ks = 0; ks < 2; ++ks){
    const int m = wave*16 + (lane & 15);
    #pragma unroll
    for (int j = 0; j < 8; ++j){
      const int k = ks*32 + ((lane >> 4) << 4) + (j << 1);
      aq[ks].u[j] = *(const unsigned int*)&Qs[m*64 + k];
    }
  }

  float mrow[8], lrow[8];
  v8f oacc[4] = {};
  #pragma unroll
  for (int r = 0; r < 8; ++r){ mrow[r] = -1e30f; lrow[r] = 0.f; }
  unsigned short* pp = &Ps[wave][0];
  const float scale = 0.125f;           // 1/sqrt(64)

  for (int kb = 0; kb < SEQ/64; ++kb){
    { // stage K [key][d] and V transposed [d][key]
      const int key = tid >> 2;
      const int doff = (tid & 3) * 16;
      const unsigned short* ksrc = qkv + (size_t)(b*SEQ + kb*64 + key)*RS + H   + h*HD + doff;
      const unsigned short* vsrc = qkv + (size_t)(b*SEQ + kb*64 + key)*RS + 2*H + h*HD + doff;
      uint4* kdst = (uint4*)&Ks[key*64 + doff];
      const uint4* k4 = (const uint4*)ksrc;
      kdst[0] = k4[0]; kdst[1] = k4[1];
      union { uint4 q[2]; unsigned short s[16]; } vt;
      const uint4* v4 = (const uint4*)vsrc;
      vt.q[0] = v4[0]; vt.q[1] = v4[1];
      #pragma unroll
      for (int j = 0; j < 16; ++j) Vs[(doff + j)*64 + key] = vt.s[j];
    }
    __syncthreads();

    v8f sacc[4] = {};                              // scores 16x64
    #pragma unroll
    for (int ni = 0; ni < 4; ++ni){
      #pragma unroll
      for (int ks = 0; ks < 2; ++ks){
        ABfrag bk;
        const int n   = ni*16 + (lane & 15);
        const int kst = ks*32 + ((lane >> 4) << 4);
        const uint4* p = (const uint4*)&Ks[n*64 + kst];
        bk.q[0] = p[0]; bk.q[1] = p[1];
        sacc[ni] = wmma_bf16(aq[ks], bk, sacc[ni]);
      }
    }

    float kbias[4];
    #pragma unroll
    for (int ni = 0; ni < 4; ++ni){
      const int key = kb*64 + ni*16 + (lane & 15);
      kbias[ni] = (1.f - (float)amask[b*SEQ + key]) * -10000.f;
    }

    #pragma unroll
    for (int r = 0; r < 8; ++r){                   // online softmax per row
      float mx = -1e30f;
      #pragma unroll
      for (int ni = 0; ni < 4; ++ni){
        const float t = sacc[ni][r]*scale + kbias[ni];
        sacc[ni][r] = t;
        mx = fmaxf(mx, t);
      }
      #pragma unroll
      for (int off = 1; off < 16; off <<= 1) mx = fmaxf(mx, __shfl_xor(mx, off, 32));
      const float mnew = fmaxf(mrow[r], mx);
      const float corr = __expf(mrow[r] - mnew);
      float ls = 0.f;
      #pragma unroll
      for (int ni = 0; ni < 4; ++ni){
        const float pe = __expf(sacc[ni][r] - mnew);
        sacc[ni][r] = pe; ls += pe;
      }
      #pragma unroll
      for (int off = 1; off < 16; off <<= 1) ls += __shfl_xor(ls, off, 32);
      lrow[r] = lrow[r]*corr + ls;
      mrow[r] = mnew;
      #pragma unroll
      for (int ni = 0; ni < 4; ++ni) oacc[ni][r] = oacc[ni][r] * corr;
    }

    #pragma unroll
    for (int ni = 0; ni < 4; ++ni)                 // P (C-layout) -> LDS row-major
      #pragma unroll
      for (int r = 0; r < 8; ++r){
        const int m = ((lane >> 4) << 3) + r;
        const int n = ni*16 + (lane & 15);
        pp[m*64 + n] = f32_to_bf16(sacc[ni][r]);
      }

    ABfrag ap[2];                                  // P as A-frags
    #pragma unroll
    for (int ks = 0; ks < 2; ++ks){
      const int m = lane & 15;
      #pragma unroll
      for (int j = 0; j < 8; ++j){
        const int k = ks*32 + ((lane >> 4) << 4) + (j << 1);
        ap[ks].u[j] = *(const unsigned int*)&pp[m*64 + k];
      }
    }
    #pragma unroll
    for (int nd = 0; nd < 4; ++nd){                // O += P x V
      #pragma unroll
      for (int ks = 0; ks < 2; ++ks){
        ABfrag bv;
        const int d   = nd*16 + (lane & 15);
        const int kst = ks*32 + ((lane >> 4) << 4);
        const uint4* p = (const uint4*)&Vs[d*64 + kst];
        bv.q[0] = p[0]; bv.q[1] = p[1];
        oacc[nd] = wmma_bf16(ap[ks], bv, oacc[nd]);
      }
    }
    __syncthreads();
  }

  #pragma unroll
  for (int nd = 0; nd < 4; ++nd){
    #pragma unroll
    for (int r = 0; r < 8; ++r){
      const int q = qblk*128 + wave*16 + ((lane >> 4) << 3) + r;
      const int d = nd*16 + (lane & 15);
      const float vv = oacc[nd][r] / lrow[r];
      ctx[(size_t)(b*SEQ + q)*H + h*HD + d] = f32_to_bf16(vv);
    }
  }
}

// ---------------------------------------------------------------------------
// Pooler + intent head (tiny)
// ---------------------------------------------------------------------------
__global__ __launch_bounds__(256) void k_pool_intent(
    const float* __restrict__ x, const float* __restrict__ pw, const float* __restrict__ pb,
    const float* __restrict__ iw, const float* __restrict__ ib, float* __restrict__ out)
{
  __shared__ float cls[H];
  __shared__ float pooled[H];
  const int b = blockIdx.x, tid = threadIdx.x;
  for (int i = tid; i < H; i += 256) cls[i] = x[(size_t)b*SEQ*H + i];
  __syncthreads();
  for (int j = tid; j < H; j += 256){
    float a = pb[j];
    for (int k = 0; k < H; ++k) a += cls[k] * pw[(size_t)k*H + j];
    pooled[j] = tanhf(a);
  }
  __syncthreads();
  if (tid < NINTENT){
    float a = ib[tid];
    for (int k = 0; k < H; ++k) a += pooled[k] * iw[k*NINTENT + tid];
    out[b*NINTENT + tid] = a;
  }
}

// slot_logits[b, n, m] = x[b, mapping[b,m]] . slot_w[:, n] + slot_b[n]
__global__ __launch_bounds__(128) void k_slot(
    const float* __restrict__ x, const int* __restrict__ mapping,
    const float* __restrict__ sw, const float* __restrict__ sb,
    float* __restrict__ out)
{
  __shared__ float row[H];
  const int m = blockIdx.x, b = blockIdx.y, tid = threadIdx.x;
  const int src = mapping[b*MMAP + m];
  const float* xr = x + ((size_t)b*SEQ + (size_t)src) * H;
  for (int i = tid; i < H; i += 128) row[i] = xr[i];
  __syncthreads();
  float a = sb[tid];
  for (int k = 0; k < H; ++k) a += row[k] * sw[k*NSLOT + tid];
  out[(size_t)b*NSLOT*MMAP + (size_t)tid*MMAP + m] = a;
}

// ---------------------------------------------------------------------------
// Launcher
// ---------------------------------------------------------------------------
extern "C" void kernel_launch(void* const* d_in, const int* in_sizes, int n_in,
                              void* d_out, int out_size, void* d_ws, size_t ws_size,
                              hipStream_t stream)
{
  const int*   ids   = (const int*)d_in[0];
  const int*   amask = (const int*)d_in[1];
  const int*   mapping = (const int*)d_in[2];
  // d_in[3] = lens (unused by the reference)
  const float* wemb  = (const float*)d_in[4];
  const float* pemb  = (const float*)d_in[5];
  const float* temb  = (const float*)d_in[6];
  const float* eln_s = (const float*)d_in[7];
  const float* eln_b = (const float*)d_in[8];
  const float* qkv_w = (const float*)d_in[9];
  const float* qkv_b = (const float*)d_in[10];
  const float* aow   = (const float*)d_in[11];
  const float* aob   = (const float*)d_in[12];
  const float* ln1s  = (const float*)d_in[13];
  const float* ln1b  = (const float*)d_in[14];
  const float* fw1   = (const float*)d_in[15];
  const float* fb1   = (const float*)d_in[16];
  const float* fw2   = (const float*)d_in[17];
  const float* fb2   = (const float*)d_in[18];
  const float* ln2s  = (const float*)d_in[19];
  const float* ln2b  = (const float*)d_in[20];
  const float* pw    = (const float*)d_in[21];
  const float* pb    = (const float*)d_in[22];
  const float* iw    = (const float*)d_in[23];
  const float* ib    = (const float*)d_in[24];
  const float* sw    = (const float*)d_in[25];
  const float* sb    = (const float*)d_in[26];
  float* out = (float*)d_out;
  (void)in_sizes; (void)n_in; (void)out_size; (void)ws_size;

  size_t off = 0;
  auto alloc = [&](size_t bytes) -> void* {
    void* p = (char*)d_ws + off;
    off += (bytes + 255) & ~(size_t)255;
    return p;
  };
  unsigned short* wt   = (unsigned short*)alloc((size_t)H*FF*2);      // reusable [N][K] bf16
  float*          xf   = (float*)         alloc((size_t)MROWS*H*4);
  unsigned short* xb   = (unsigned short*)alloc((size_t)MROWS*H*2);
  unsigned short* qkvb = (unsigned short*)alloc((size_t)MROWS*3*H*2);
  unsigned short* ctxb = (unsigned short*)alloc((size_t)MROWS*H*2);
  unsigned short* h1b  = (unsigned short*)alloc((size_t)MROWS*FF*2);
  float*          y32  = (float*)         alloc((size_t)MROWS*H*4);

  k_embed_ln<<<MROWS, 256, 0, stream>>>(ids, wemb, pemb, temb, eln_s, eln_b, xf, xb);

  for (int l = 0; l < LAYERS; ++l){
    // QKV projection: [8192,768] x [768,2304]
    k_transpose_bf16<<<(H*3*H + 255)/256, 256, 0, stream>>>(qkv_w + (size_t)l*H*3*H, wt, H, 3*H);
    k_gemm<EPI_BF16><<<dim3(3*H/128, MROWS/128), 256, 0, stream>>>(
        xb, wt, qkv_b + (size_t)l*3*H, qkvb, nullptr, 3*H, H);

    // attention
    k_attention<<<dim3(SEQ/128, NH, BATCH), 256, 0, stream>>>(qkvb, amask, ctxb);

    // output projection: [8192,768] x [768,768] -> fp32
    k_transpose_bf16<<<(H*H + 255)/256, 256, 0, stream>>>(aow + (size_t)l*H*H, wt, H, H);
    k_gemm<EPI_F32><<<dim3(H/128, MROWS/128), 256, 0, stream>>>(
        ctxb, wt, aob + (size_t)l*H, nullptr, y32, H, H);
    k_resid_ln<<<MROWS, 256, 0, stream>>>(xf, y32, ln1s + (size_t)l*H, ln1b + (size_t)l*H, xf, xb);

    // FFN1 + GELU: [8192,768] x [768,3072] -> bf16
    k_transpose_bf16<<<(H*FF + 255)/256, 256, 0, stream>>>(fw1 + (size_t)l*H*FF, wt, H, FF);
    k_gemm<EPI_GELU><<<dim3(FF/128, MROWS/128), 256, 0, stream>>>(
        xb, wt, fb1 + (size_t)l*FF, h1b, nullptr, FF, H);

    // FFN2: [8192,3072] x [3072,768] -> fp32
    k_transpose_bf16<<<(FF*H + 255)/256, 256, 0, stream>>>(fw2 + (size_t)l*FF*H, wt, FF, H);
    k_gemm<EPI_F32><<<dim3(H/128, MROWS/128), 256, 0, stream>>>(
        h1b, wt, fb2 + (size_t)l*H, nullptr, y32, H, FF);
    k_resid_ln<<<MROWS, 256, 0, stream>>>(xf, y32, ln2s + (size_t)l*H, ln2b + (size_t)l*H, xf, xb);
  }

  k_pool_intent<<<BATCH, 256, 0, stream>>>(xf, pw, pb, iw, ib, out);
  k_slot<<<dim3(MMAP, BATCH), 128, 0, stream>>>(xf, mapping, sw, sb, out + BATCH*NINTENT);
}
